// DensityEdgeProjection_1580547975071
// MI455X (gfx1250) — compile-verified, matrix operand order checked by reading, not run on hardware
//
#include <hip/hip_runtime.h>

typedef _Float16 half_t;
typedef half_t v16h __attribute__((ext_vector_type(16)));
typedef float  v8f  __attribute__((ext_vector_type(8)));

#define LDH 264   // f16 row stride for 64x256 tiles in LDS
#define LDP 72    // f16 row stride for per-head 32x64 softmax scratch
#define NWG 256   // phase-1 workgroups (split-K partials)
#define NT  4608  // 2*384*(384/64) kv tiles

__device__ __forceinline__ float silu_f(float x) {
  // x * sigmoid(x) with v_rcp_f32 instead of the correctly-rounded div expansion
  return x * __builtin_amdgcn_rcpf(1.f + __expf(-x));
}

__device__ __forceinline__ v8f wmma16(v16h a, v16h b, v8f c) {
  return __builtin_amdgcn_wmma_f32_16x16x32_f16(false, a, false, b, (short)0, c,
                                                false, false);
}

// A-fragment (16x32 f16, row-major source). Per ISA 7.12.2: lane m=lane&15,
// VGPR v holds K = (v<4?0:16) + 2*(v&3) (+8 for lanes 16-31). Pairs contiguous.
__device__ __forceinline__ v16h load_a_frag(const half_t* p, int ld, int row0,
                                            int k0, int lane) {
  int r = lane & 15, hf = lane >> 4;
  const half_t* rp = p + (size_t)(row0 + r) * ld + k0 + 8 * hf;
  union { v16h v; unsigned u[8]; } f;
#pragma unroll
  for (int v = 0; v < 8; ++v) {
    int k = ((v & 4) ? 16 : 0) + 2 * (v & 3);
    f.u[v] = *(const unsigned*)(rp + k);
  }
  return f.v;
}

// B-fragment where B's K dimension is the row index of a row-major buffer
// (i.e. column access): lane n holds column col0+n, K-interleaved.
__device__ __forceinline__ v16h load_b_col_frag(const half_t* p, int ld, int k0,
                                                int col0, int lane) {
  int n = lane & 15, hf = lane >> 4;
  union { v16h v; half_t h[16]; } f;
#pragma unroll
  for (int v = 0; v < 8; ++v) {
    int k = ((v & 4) ? 16 : 0) + 2 * (v & 3) + 8 * hf;
    f.h[2 * v]     = p[(size_t)(k0 + k) * ld + col0 + n];
    f.h[2 * v + 1] = p[(size_t)(k0 + k + 1) * ld + col0 + n];
  }
  return f.v;
}

// 64x256 tile GEMM: sOut = act(sIn @ W + bias), W given pre-transposed f16
// (Wt[n][k]).  8 waves: wave = (colhalf<<2)|rowblock. MODE 0 = SiLU, 1 = linear.
// Low-pressure form: 2 column blocks per pass sharing one A fragment that is
// re-read from LDS each k-step (A re-read is 2x ds_load_b128 -- cheap).
template <int MODE>
__device__ __forceinline__ void gemm_tile(const half_t* __restrict__ sIn,
                                          const half_t* __restrict__ Wt,
                                          const float* __restrict__ bias,
                                          half_t* __restrict__ sOut, int lane,
                                          int wid) {
  const int rb = wid & 3;
  const int colBase = (wid >> 2) * 128;
  const int n = lane & 15, hf = lane >> 4;
#pragma unroll 1
  for (int cp = 0; cp < 4; ++cp) {
    const int col0 = colBase + cp * 32;
    v8f acc0 = {}, acc1 = {};
#pragma unroll 2
    for (int ks = 0; ks < 8; ++ks) {
      v16h A  = load_a_frag(sIn, LDH, rb * 16, ks * 32, lane);
      v16h B0 = load_a_frag(Wt, 256, col0, ks * 32, lane);
      v16h B1 = load_a_frag(Wt, 256, col0 + 16, ks * 32, lane);
      acc0 = wmma16(A, B0, acc0);
      acc1 = wmma16(A, B1, acc1);
    }
    const float bb0 = bias[col0 + n];
    const float bb1 = bias[col0 + 16 + n];
#pragma unroll
    for (int v = 0; v < 8; ++v) {
      float x0 = acc0[v] + bb0;
      float x1 = acc1[v] + bb1;
      if (MODE == 0) {
        x0 = silu_f(x0);
        x1 = silu_f(x1);
      }
      int row = rb * 16 + v + 8 * hf;
      sOut[(size_t)row * LDH + col0 + n] = (half_t)x0;
      sOut[(size_t)row * LDH + col0 + 16 + n] = (half_t)x1;
    }
  }
}

// ---------------- phase 0 kernels (tiny precompute) ----------------

__global__ void k_h1(const int* __restrict__ Z, const int* __restrict__ L,
                     const int* __restrict__ Mm, const float* __restrict__ elem,
                     const float* __restrict__ lemb, const float* __restrict__ memb,
                     const float* __restrict__ w1, const float* __restrict__ b1,
                     float* __restrict__ h1) {
  for (int idx = blockIdx.x * blockDim.x + threadIdx.x; idx < 384 * 128;
       idx += gridDim.x * blockDim.x) {
    int b = idx >> 7, c = idx & 127;
    const float* we = elem + Z[b] * 32;
    const float* wl = lemb + L[b] * 16;
    int msh = Mm[b] + 2; msh = msh < 0 ? 0 : (msh > 4 ? 4 : msh);
    const float* wm = memb + msh * 16;
    float acc = b1[c];
    for (int k = 0; k < 32; ++k) acc += we[k] * w1[k * 128 + c];
    for (int k = 0; k < 16; ++k) acc += wl[k] * w1[(32 + k) * 128 + c];
    for (int k = 0; k < 16; ++k) acc += wm[k] * w1[(48 + k) * 128 + c];
    h1[idx] = silu_f(acc);
  }
}

__global__ void k_orb(const float* __restrict__ h1, const float* __restrict__ w2,
                      const float* __restrict__ b2, float* __restrict__ orb) {
  for (int idx = blockIdx.x * blockDim.x + threadIdx.x; idx < 384 * 128;
       idx += gridDim.x * blockDim.x) {
    int b = idx >> 7, c = idx & 127;
    float acc = b2[c];
    for (int k = 0; k < 128; ++k) acc += h1[b * 128 + k] * w2[k * 128 + c];
    orb[idx] = acc;
  }
}

// a_i = orb @ enc_w1[0:128] + enc_b1 ; a_j = orb @ enc_w1[128:256]
__global__ void k_aij(const float* __restrict__ orb, const float* __restrict__ ew1,
                      const float* __restrict__ eb1, float* __restrict__ ai,
                      float* __restrict__ aj) {
  for (int idx = blockIdx.x * blockDim.x + threadIdx.x; idx < 2 * 384 * 256;
       idx += gridDim.x * blockDim.x) {
    int which = idx / 98304;
    int r = idx - which * 98304;
    int b = r >> 8, c = r & 255;
    const float* wb = ew1 + (which ? 128 * 256 : 0);
    float acc = which ? 0.f : eb1[c];
    for (int k = 0; k < 128; ++k) acc += orb[b * 128 + k] * wb[k * 256 + c];
    (which ? aj : ai)[b * 256 + c] = acc;
  }
}

// q = LN(query_tokens); Qh[head][q][d] = (q@wq+bq)*rsqrt(32), f16
__global__ __launch_bounds__(256) void k_q(const float* __restrict__ qt,
                                           const float* __restrict__ lnw,
                                           const float* __restrict__ lnb,
                                           const float* __restrict__ wq,
                                           const float* __restrict__ bq,
                                           half_t* __restrict__ Qh) {
  __shared__ float qn[32 * 256];
  __shared__ float red[64];
  int tid = threadIdx.x;
  if (tid < 32) {
    float s1 = 0.f, s2 = 0.f;
    for (int c = 0; c < 256; ++c) { float x = qt[tid * 256 + c]; s1 += x; s2 += x * x; }
    float mu = s1 * (1.f / 256.f);
    float var = s2 * (1.f / 256.f) - mu * mu;
    red[tid * 2] = mu;
    red[tid * 2 + 1] = rsqrtf(var + 1e-5f);
  }
  __syncthreads();
  for (int idx = tid; idx < 8192; idx += 256) {
    int r = idx >> 8, c = idx & 255;
    qn[idx] = (qt[idx] - red[r * 2]) * red[r * 2 + 1] * lnw[c] + lnb[c];
  }
  __syncthreads();
  for (int idx = tid; idx < 8192; idx += 256) {
    int r = idx >> 8, c = idx & 255;
    float acc = bq[c];
    for (int k = 0; k < 256; ++k) acc += qn[r * 256 + k] * wq[k * 256 + c];
    acc *= 0.17677669529663687f;  // 1/sqrt(dh=32)
    Qh[(size_t)(c >> 5) * 1024 + r * 32 + (c & 31)] = (half_t)acc;
  }
}

// transpose + f32->f16 the four 256x256 weight matrices
__global__ void k_tr(const float* __restrict__ s0, const float* __restrict__ s1,
                     const float* __restrict__ s2, const float* __restrict__ s3,
                     half_t* __restrict__ d0, half_t* __restrict__ d1,
                     half_t* __restrict__ d2, half_t* __restrict__ d3) {
  for (int idx = blockIdx.x * blockDim.x + threadIdx.x; idx < 4 * 65536;
       idx += gridDim.x * blockDim.x) {
    int m = idx >> 16, e = idx & 65535;
    int k = e >> 8, n = e & 255;
    const float* s = (m == 0) ? s0 : (m == 1) ? s1 : (m == 2) ? s2 : s3;
    half_t* d = (m == 0) ? d0 : (m == 1) ? d1 : (m == 2) ? d2 : d3;
    d[(size_t)n * 256 + k] = (half_t)s[e];
  }
}

// ---------------- phase 1: fused kv pipeline + flash attention ----------------

__global__ __launch_bounds__(256, 1) void k_fused(
    const float* __restrict__ rho_r, const float* __restrict__ rho_i,
    const float* __restrict__ a_i, const float* __restrict__ a_j,
    const float* __restrict__ enc_w1,
    const half_t* __restrict__ W2t, const float* __restrict__ b2,
    const half_t* __restrict__ W3t, const float* __restrict__ b3,
    const float* __restrict__ lnw, const float* __restrict__ lnb,
    const half_t* __restrict__ Wkt, const float* __restrict__ bk_,
    const half_t* __restrict__ Wvt, const float* __restrict__ bv_,
    const half_t* __restrict__ Qh,
    float* __restrict__ pm, float* __restrict__ pl, float* __restrict__ pO) {
  // 36864 (h/kv buffer, aliased later as P scratch 8*32*LDP) + K + V + alpha + red
  __shared__ __align__(16) unsigned char smem[36864 + 33792 + 33792 + 1024 + 2048];
  half_t* sA = (half_t*)smem;
  half_t* sB = (half_t*)(smem + 36864);
  half_t* sC = (half_t*)(smem + 36864 + 33792);
  float* sAl = (float*)(smem + 36864 + 33792 + 33792);
  float* sRed = sAl + 256;

  const int tid = threadIdx.x;
  const int lane = tid & 31;
  const int wid = tid >> 5;
  const int hd = wid;  // one head per wave

  float m_ = -1e30f, l_ = 0.f;
  v8f O00 = {}, O01 = {}, O10 = {}, O11 = {};

  const float* wr = enc_w1 + 256 * 256;
  const float* wim = enc_w1 + 257 * 256;

  for (int t = blockIdx.x; t < NT; t += gridDim.x) {
    int s = t / (384 * 6);
    int r1 = t - s * (384 * 6);
    int i = r1 / 6;
    int jb = (r1 - i * 6) * 64;

    // ---- h tile: silu(a_i + a_j + rho_r*w_r + rho_i*w_im) -> f16 in sA ----
    {
      int r = tid >> 2;
      int c0 = (tid & 3) * 64;
      int j = jb + r;
      size_t ridx = (size_t)(s * 384 + i) * 384 + j;
      float rr = rho_r[ridx];
      float ri = rho_i[ridx];
      const float* ai = a_i + (size_t)i * 256;  // includes enc_b1
      const float* aj = a_j + (size_t)j * 256;
      half_t* dst = sA + (size_t)r * LDH;
      for (int cc = 0; cc < 64; ++cc) {
        int c = c0 + cc;
        float x = ai[c] + aj[c] + rr * wr[c] + ri * wim[c];
        dst[c] = (half_t)silu_f(x);
      }
    }
    __syncthreads();
    gemm_tile<0>(sA, W2t, b2, sB, lane, wid);  // silu(h @ W2 + b2)
    __syncthreads();
    gemm_tile<1>(sB, W3t, b3, sA, lane, wid);  // @ W3 + b3 (raw)
    __syncthreads();
    // ---- per-row layernorm on sA ----
    {
      int r = tid >> 2, sub = tid & 3;
      const half_t* rp = sA + (size_t)r * LDH + sub * 64;
      float s1 = 0.f, s2 = 0.f;
      for (int cc = 0; cc < 64; ++cc) { float x = (float)rp[cc]; s1 += x; s2 += x * x; }
      sRed[(r * 4 + sub) * 2] = s1;
      sRed[(r * 4 + sub) * 2 + 1] = s2;
    }
    __syncthreads();
    {
      int r = tid >> 2, sub = tid & 3;
      float s1 = 0.f, s2 = 0.f;
      for (int u = 0; u < 4; ++u) { s1 += sRed[(r * 4 + u) * 2]; s2 += sRed[(r * 4 + u) * 2 + 1]; }
      float mu = s1 * (1.f / 256.f);
      float var = s2 * (1.f / 256.f) - mu * mu;
      float rs = rsqrtf(var + 1e-5f);
      half_t* rp = sA + (size_t)r * LDH + sub * 64;
      int c0 = sub * 64;
      for (int cc = 0; cc < 64; ++cc) {
        float x = ((float)rp[cc] - mu) * rs * lnw[c0 + cc] + lnb[c0 + cc];
        rp[cc] = (half_t)x;
      }
    }
    __syncthreads();
    gemm_tile<1>(sA, Wkt, bk_, sB, lane, wid);  // K
    gemm_tile<1>(sA, Wvt, bv_, sC, lane, wid);  // V
    __syncthreads();

    // ---- attention: sA region is now free -> per-head P scratch ----
    half_t* sP = (half_t*)smem + (size_t)hd * 32 * LDP;
    {
      // Q fragments reloaded per tile (2 KB in L2) to keep them out of the
      // GEMM sections' live range.
      v16h qf0 = load_a_frag(Qh + (size_t)hd * 1024, 32, 0, 0, lane);
      v16h qf1 = load_a_frag(Qh + (size_t)hd * 1024, 32, 16, 0, lane);
      v8f z = {};
#pragma unroll 1
      for (int nb = 0; nb < 4; ++nb) {
        v16h Kf = load_a_frag(sB, LDH, nb * 16, hd * 32, lane);  // K rows as B
        v8f s0 = wmma16(qf0, Kf, z);
        v8f s1 = wmma16(qf1, Kf, z);
        int n = lane & 15, hf = lane >> 4;
#pragma unroll
        for (int v = 0; v < 8; ++v) {
          sP[(v + 8 * hf) * LDP + nb * 16 + n] = (half_t)s0[v];
          sP[(16 + v + 8 * hf) * LDP + nb * 16 + n] = (half_t)s1[v];
        }
      }
    }
    __syncthreads();
    {  // streaming softmax: lane == query row
      half_t* rowp = sP + lane * LDP;
      float mx = m_;
      for (int c = 0; c < 64; ++c) mx = fmaxf(mx, (float)rowp[c]);
      float alpha = __expf(m_ - mx);
      float sum = 0.f;
      for (int c = 0; c < 64; ++c) {
        float p = __expf((float)rowp[c] - mx);
        sum += p;
        rowp[c] = (half_t)p;
      }
      l_ = l_ * alpha + sum;
      m_ = mx;
      sAl[hd * 32 + lane] = alpha;
    }
    __syncthreads();
    {
      int hf = lane >> 4;
#pragma unroll
      for (int v = 0; v < 8; ++v) {
        float a0 = sAl[hd * 32 + v + 8 * hf];
        float a1 = sAl[hd * 32 + 16 + v + 8 * hf];
        O00[v] *= a0; O01[v] *= a0;
        O10[v] *= a1; O11[v] *= a1;
      }
#pragma unroll 1
      for (int kb = 0; kb < 2; ++kb) {
        v16h P0 = load_a_frag(sP, LDP, 0, kb * 32, lane);
        v16h P1 = load_a_frag(sP, LDP, 16, kb * 32, lane);
        v16h V0 = load_b_col_frag(sC, LDH, kb * 32, hd * 32, lane);
        v16h V1 = load_b_col_frag(sC, LDH, kb * 32, hd * 32 + 16, lane);
        O00 = wmma16(P0, V0, O00);
        O01 = wmma16(P0, V1, O01);
        O10 = wmma16(P1, V0, O10);
        O11 = wmma16(P1, V1, O11);
      }
    }
    __syncthreads();
  }

  // ---- write split-K partials ----
  {
    size_t base = (size_t)blockIdx.x * 256 + hd * 32 + lane;
    pm[base] = m_;
    pl[base] = l_;
    int n = lane & 15, hf = lane >> 4;
    float* Ob = pO + ((size_t)blockIdx.x * 8 + hd) * 1024;
#pragma unroll
    for (int v = 0; v < 8; ++v) {
      int q0 = v + 8 * hf;
      Ob[q0 * 32 + n] = O00[v];
      Ob[q0 * 32 + 16 + n] = O01[v];
      Ob[(16 + q0) * 32 + n] = O10[v];
      Ob[(16 + q0) * 32 + 16 + n] = O11[v];
    }
  }
}

// ---------------- phase 2: combine partials + output MLP ----------------

__global__ __launch_bounds__(256) void k_out(
    const float* __restrict__ pm, const float* __restrict__ pl,
    const float* __restrict__ pO, const float* __restrict__ wo,
    const float* __restrict__ bo, const float* __restrict__ w1,
    const float* __restrict__ b1, const float* __restrict__ w2,
    const float* __restrict__ b2, float* __restrict__ outp, int nwg) {
  __shared__ float bufA[32 * 256];
  __shared__ float bufB[32 * 256];
  int tid = threadIdx.x;
  {  // flash-attention combine: thread = (head, query)
    int hd = tid >> 5, q = tid & 31;
    float M = -1e30f;
    for (int p = 0; p < nwg; ++p) M = fmaxf(M, pm[p * 256 + hd * 32 + q]);
    float denom = 0.f;
    float o[32];
#pragma unroll
    for (int n = 0; n < 32; ++n) o[n] = 0.f;
    for (int p = 0; p < nwg; ++p) {
      float w = __expf(pm[p * 256 + hd * 32 + q] - M);
      denom += pl[p * 256 + hd * 32 + q] * w;
      const float* Op = pO + ((size_t)p * 8 + hd) * 1024 + q * 32;
#pragma unroll
      for (int n = 0; n < 32; ++n) o[n] += Op[n] * w;
    }
    float inv = 1.f / denom;
    for (int n = 0; n < 32; ++n) bufA[q * 256 + hd * 32 + n] = o[n] * inv;
  }
  __syncthreads();
  {  // attended = ctx @ wo + bo
    int r = tid >> 3, c0 = (tid & 7) * 32;
    for (int cc = 0; cc < 32; ++cc) {
      int c = c0 + cc;
      float acc = bo[c];
      for (int k = 0; k < 256; ++k) acc += bufA[r * 256 + k] * wo[k * 256 + c];
      bufB[r * 256 + c] = acc;
    }
  }
  __syncthreads();
  {  // t1 = silu(attended @ out_w1 + out_b1)
    int r = tid >> 3, c0 = (tid & 7) * 32;
    for (int cc = 0; cc < 32; ++cc) {
      int c = c0 + cc;
      float acc = b1[c];
      for (int k = 0; k < 256; ++k) acc += bufB[r * 256 + k] * w1[k * 256 + c];
      bufA[r * 256 + c] = silu_f(acc);
    }
  }
  __syncthreads();
  {  // out = t1 @ out_w2 + out_b2
    int r = tid >> 3, c0 = (tid & 7) * 32;
    for (int cc = 0; cc < 32; ++cc) {
      int c = c0 + cc;
      float acc = b2[c];
      for (int k = 0; k < 256; ++k) acc += bufA[r * 256 + k] * w2[k * 256 + c];
      outp[r * 256 + c] = acc;
    }
  }
}

extern "C" void kernel_launch(void* const* d_in, const int* in_sizes, int n_in,
                              void* d_out, int out_size, void* d_ws,
                              size_t ws_size, hipStream_t stream) {
  (void)in_sizes; (void)n_in; (void)out_size; (void)ws_size;
  const float* rho_r = (const float*)d_in[0];
  const float* rho_i = (const float*)d_in[1];
  const int* Z = (const int*)d_in[2];
  const int* L = (const int*)d_in[3];
  const int* Mm = (const int*)d_in[4];
  const float* elem = (const float*)d_in[5];
  const float* lemb = (const float*)d_in[6];
  const float* memb = (const float*)d_in[7];
  const float* pw1 = (const float*)d_in[8];
  const float* pb1 = (const float*)d_in[9];
  const float* pw2 = (const float*)d_in[10];
  const float* pb2 = (const float*)d_in[11];
  const float* ew1 = (const float*)d_in[12];
  const float* eb1 = (const float*)d_in[13];
  const float* ew2 = (const float*)d_in[14];
  const float* eb2 = (const float*)d_in[15];
  const float* ew3 = (const float*)d_in[16];
  const float* eb3 = (const float*)d_in[17];
  const float* qt = (const float*)d_in[18];
  const float* lqw = (const float*)d_in[19];
  const float* lqb = (const float*)d_in[20];
  const float* lkw = (const float*)d_in[21];
  const float* lkb = (const float*)d_in[22];
  const float* wq = (const float*)d_in[23];
  const float* bq = (const float*)d_in[24];
  const float* wk = (const float*)d_in[25];
  const float* bk = (const float*)d_in[26];
  const float* wv = (const float*)d_in[27];
  const float* bv = (const float*)d_in[28];
  const float* wo = (const float*)d_in[29];
  const float* bo = (const float*)d_in[30];
  const float* ow1 = (const float*)d_in[31];
  const float* ob1 = (const float*)d_in[32];
  const float* ow2 = (const float*)d_in[33];
  const float* ob2 = (const float*)d_in[34];

  char* ws = (char*)d_ws;
  size_t off = 0;
  auto alloc = [&](size_t bytes) -> void* {
    void* p = ws + off;
    off += (bytes + 255) & ~(size_t)255;
    return p;
  };
  float* h1 = (float*)alloc(384 * 128 * 4);
  float* orb = (float*)alloc(384 * 128 * 4);
  float* ai = (float*)alloc(384 * 256 * 4);
  float* aj = (float*)alloc(384 * 256 * 4);
  half_t* Qh = (half_t*)alloc(8 * 32 * 32 * 2);
  half_t* W2t = (half_t*)alloc(65536 * 2);
  half_t* W3t = (half_t*)alloc(65536 * 2);
  half_t* Wkt = (half_t*)alloc(65536 * 2);
  half_t* Wvt = (half_t*)alloc(65536 * 2);
  float* pm = (float*)alloc((size_t)NWG * 256 * 4);
  float* pl = (float*)alloc((size_t)NWG * 256 * 4);
  float* pO = (float*)alloc((size_t)NWG * 8 * 1024 * 4);

  k_h1<<<48, 256, 0, stream>>>(Z, L, Mm, elem, lemb, memb, pw1, pb1, h1);
  k_orb<<<48, 256, 0, stream>>>(h1, pw2, pb2, orb);
  k_aij<<<192, 256, 0, stream>>>(orb, ew1, eb1, ai, aj);
  k_q<<<1, 256, 0, stream>>>(qt, lqw, lqb, wq, bq, Qh);
  k_tr<<<256, 256, 0, stream>>>(ew2, ew3, wk, wv, W2t, W3t, Wkt, Wvt);
  k_fused<<<NWG, 256, 0, stream>>>(rho_r, rho_i, ai, aj, ew1, W2t, eb2, W3t,
                                   eb3, lkw, lkb, Wkt, bk, Wvt, bv, Qh, pm, pl,
                                   pO);
  k_out<<<1, 256, 0, stream>>>(pm, pl, pO, wo, bo, ow1, ob1, ow2, ob2,
                               (float*)d_out, NWG);
}
